// GraphTCNEncoder_70729521430860
// MI455X (gfx1250) — matrix operand order repaired and checked by reading, not compiled, not run on hardware
//
#include <hip/hip_runtime.h>

typedef __attribute__((ext_vector_type(16))) _Float16 v16h;
typedef __attribute__((ext_vector_type(8)))  _Float16 v8h;
typedef __attribute__((ext_vector_type(8)))  float    v8f;

__device__ __forceinline__ float lrelu(float x){ return x > 0.f ? x : 0.01f*x; }
__device__ __forceinline__ float sigmf(float x){ return 1.f/(1.f+__expf(-x)); }

// ------------------------------------------------------------------
// zero a float region
__global__ void kzero(float* __restrict__ p, int n){
  int i = blockIdx.x*256 + threadIdx.x;
  if (i < n) p[i] = 0.f;
}

// ------------------------------------------------------------------
// Pre-swizzle conv weights (f32 [64][Cin][5] or skip [64][Cin][1]) into
// WMMA A-fragment order: dst[(coTile*NCK+ck)*512 + lane*16 + j], f16.
// k = r*CIN + c ;  A layout: k = ck*32 + 16*(j/8) + 8*(lane/16) + (j%8)
template<int CIN, bool SKIPMODE>
__global__ void kprepw(const float* __restrict__ W, _Float16* __restrict__ dst){
  constexpr int K5  = 5*CIN;
  constexpr int NCK = (K5 + 31)/32;
  int idx = blockIdx.x*256 + threadIdx.x;
  if (idx >= 2048*NCK) return;
  int j    = idx & 15;
  int lane = (idx >> 4) & 31;
  int ck   = (idx >> 9) % NCK;
  int co   = idx / (512*NCK);
  int cout = co*16 + (lane & 15);
  int k    = ck*32 + ((j >> 3) << 4) + ((lane >> 4) << 3) + (j & 7);
  float v = 0.f;
  if (k < K5){
    int r = k / CIN; int c = k - r*CIN;
    v = SKIPMODE ? (r == 4 ? W[cout*CIN + c] : 0.f)
                 : W[(cout*CIN + c)*5 + r];
  }
  dst[idx] = (_Float16)v;
}

// ------------------------------------------------------------------
// GAT heads + channel concat. Writes xcur f16 [1024][36][512].
// One wave per (b,t); lanes 0-15 = head0 nodes, 16-31 = head1 nodes.
__global__ __launch_bounds__(256) void kgat(
    const float* __restrict__ x,
    const float* Wsp0,const float* bsp0,const float* Wn0,const float* bn0,const float* Wa0,const float* ba0,
    const float* Wsp1,const float* bsp1,const float* Wn1,const float* bn1,const float* Wa1,const float* ba1,
    _Float16* __restrict__ xc)
{
  __shared__ float nds[8][2][16][16];
  __shared__ float pj [8][2][16];
  __shared__ float uu [8][2][16];
  int tid = threadIdx.x, wv = tid >> 5, lane = tid & 31;
  int h = lane >> 4, n = lane & 15;
  int b = blockIdx.x, t = blockIdx.y*8 + wv;
  const float* Wsp = h ? Wsp1 : Wsp0;  const float* bsp = h ? bsp1 : bsp0;
  const float* Wn  = h ? Wn1  : Wn0;   const float* bn  = h ? bn1  : bn0;
  const float* Wa  = h ? Wa1  : Wa0;   const float* ba  = h ? ba1  : ba0;

  int m = b*16 + n;
  float xf[4];
  #pragma unroll
  for (int f = 0; f < 4; ++f) xf[f] = x[((size_t)m*512 + t)*4 + f];

  float u = 0.f, vsem = 0.f;
  #pragma unroll
  for (int e = 0; e < 16; ++e){
    float s = bn[e];
    #pragma unroll
    for (int f = 0; f < 4; ++f) s += xf[f]*Wn[f*16 + e];
    nds[wv][h][n][e] = s;
    u += s*Wa[e];
    float se = bsp[e];
    #pragma unroll
    for (int f = 0; f < 4; ++f) se += xf[f]*(Wsp[(4+f)*16 + e] + Wsp[(8+f)*16 + e]);
    vsem += se*Wa[e];
  }
  pj[wv][h][n] = u + vsem;
  uu[wv][h][n] = u;
  if (h == 0){
    #pragma unroll
    for (int f = 0; f < 4; ++f) xc[((size_t)m*36 + f)*512 + t] = (_Float16)xf[f];
  }
  __syncthreads();

  // row i = n : softmax over j (diagonal pre-softmax value is 0)
  float bav = ba[0];
  float ui  = uu[wv][h][n];
  float pre[16], mx = 0.f;
  #pragma unroll
  for (int j = 0; j < 16; ++j){
    float pv = (j == n) ? 0.f : lrelu(ui + pj[wv][h][j] + bav);
    pre[j] = pv; mx = fmaxf(mx, pv);
  }
  float w[16], Z = 0.f;
  #pragma unroll
  for (int j = 0; j < 16; ++j){ w[j] = __expf(pre[j] - mx); Z += w[j]; }
  float inv = 1.f/Z;
  #pragma unroll
  for (int e = 0; e < 16; ++e){
    float acc = 0.f;
    #pragma unroll
    for (int j = 0; j < 16; ++j) acc += w[j]*nds[wv][h][j][e];
    acc = lrelu(acc*inv);
    xc[((size_t)m*36 + 4 + h*16 + e)*512 + t] = (_Float16)acc;
  }
}

// ------------------------------------------------------------------
// Gated dilated causal conv via WMMA. Block = 128 thr (4 waves); wave co
// computes cout tile [16co..16co+15] x 16 times.
// im2col panel built ONCE per block in LDS: col[n][k]  (n = 16 times,
// k = r*CIN+c over padded K). B fragments are then two aligned 16B LDS
// vector loads per chunk — no per-element math in the hot loop.
// Optional skip (1x1 conv folded as tap r=4); optional BN affine applied
// to input while building the panel. BN stats -> 256-partition buffer.
template<int CIN, int DIL, bool SKIP, bool AFF>
__global__ __launch_bounds__(128) void kconv(
    const _Float16* __restrict__ xin,
    const _Float16* __restrict__ Wg, const _Float16* __restrict__ Ws,
    const _Float16* __restrict__ Wk,
    const float* __restrict__ bg, const float* __restrict__ bs,
    const float* __restrict__ bk,
    const float* __restrict__ aff,          // [128]: scale[64], shift[64]
    _Float16* __restrict__ hout, _Float16* __restrict__ skout,
    float* __restrict__ stats)              // 256 x 128 floats
{
  constexpr int K5   = 5*CIN;
  constexpr int NCK  = (K5 + 31)/32;
  constexpr int KP   = NCK*32;
  constexpr int LSTR = KP + 8;              // 656B rows: 16B aligned, bank-spread
  __shared__ _Float16 col[16*LSTR];
  __shared__ float bsum[64], bsq[64];

  int tid = threadIdx.x, lane = tid & 31, co = tid >> 5;
  int m = blockIdx.y, t0 = blockIdx.x*16;
  if (tid < 64){ bsum[tid] = 0.f; bsq[tid] = 0.f; }

  // build im2col panel (zero-filled causal padding + K tail)
  for (int i = tid; i < 16*KP; i += 128){
    int n = i & 15, k = i >> 4;
    float v = 0.f;
    if (k < K5){
      int r = k / CIN; int c = k - r*CIN;
      int t = t0 + n + DIL*(r - 4);
      if (t >= 0){
        v = (float)xin[((size_t)m*CIN + c)*512 + t];
        if (AFF) v = aff[c]*v + aff[64 + c];
      }
    }
    col[n*LSTR + k] = (_Float16)v;
  }
  __syncthreads();

  int tl = lane & 15;
  int kg = (lane >> 4) << 3;                 // B fragment k-offset per half-wave
  const _Float16* bptr = &col[tl*LSTR + kg];
  v8f cgv = {}; v8f csv = {}; v8f ckv = {};
  #pragma unroll
  for (int ck = 0; ck < NCK; ++ck){
    union { v16h v; v8h h[2]; } bu;
    bu.h[0] = *(const v8h*)(bptr + ck*32);
    bu.h[1] = *(const v8h*)(bptr + ck*32 + 16);
    size_t abase = ((size_t)(co*NCK + ck)*32 + lane)*16;
    v16h ag = *(const v16h*)(Wg + abase);
    v16h as = *(const v16h*)(Ws + abase);
    cgv = __builtin_amdgcn_wmma_f32_16x16x32_f16(false, ag, false, bu.v, (short)0, cgv, false, false);
    csv = __builtin_amdgcn_wmma_f32_16x16x32_f16(false, as, false, bu.v, (short)0, csv, false, false);
    if (SKIP && (ck+1)*32 > 4*CIN){
      v16h ak = *(const v16h*)(Wk + abase);
      ckv = __builtin_amdgcn_wmma_f32_16x16x32_f16(false, ak, false, bu.v, (short)0, ckv, false, false);
    }
  }

  int tglob  = t0 + tl;
  int rowoff = (lane >> 4) << 3;
  union { v8f v; float f[8]; } ug, us, uk;
  ug.v = cgv; us.v = csv; uk.v = ckv;
  #pragma unroll
  for (int v = 0; v < 8; ++v){
    int ch = co*16 + rowoff + v;
    float g = sigmf(ug.f[v] + bg[ch]);
    float s = tanhf(us.f[v] + bs[ch]);
    float hv = g*s;
    hout[((size_t)m*64 + ch)*512 + tglob] = (_Float16)hv;
    if (SKIP) skout[((size_t)m*64 + ch)*512 + tglob] = (_Float16)(uk.f[v] + bk[ch]);
    float a = hv, q = hv*hv;
    #pragma unroll
    for (int off = 8; off; off >>= 1){
      a += __shfl_xor(a, off, 32);
      q += __shfl_xor(q, off, 32);
    }
    if (tl == 0){ atomicAdd(&bsum[ch], a); atomicAdd(&bsq[ch], q); }
  }
  __syncthreads();
  if (tid < 64){
    float* part = stats + (size_t)((blockIdx.y*32 + blockIdx.x) & 255)*128;
    atomicAdd(&part[tid], bsum[tid]);
    atomicAdd(&part[64 + tid], bsq[tid]);
  }
}

// ------------------------------------------------------------------
// BN finalize: partials -> per-channel affine (scale, shift)
__global__ void kbnfin(const float* __restrict__ stats,
                       const float* __restrict__ g, const float* __restrict__ bb,
                       float* __restrict__ fin){
  int c = threadIdx.x;
  if (c >= 64) return;
  float s = 0.f, q = 0.f;
  for (int p = 0; p < 256; ++p){ s += stats[p*128 + c]; q += stats[p*128 + 64 + c]; }
  const float cnt = 1024.f*512.f;
  float mu = s/cnt;
  float var = q/cnt - mu*mu;
  float rstd = rsqrtf(var + 1e-5f);
  float sc = g[c]*rstd;
  fin[c] = sc;
  fin[64 + c] = bb[c] - mu*sc;
}

// ------------------------------------------------------------------
// x_next = BN2(h2) + skip    (all [1024][64][512], f16)
__global__ void kapply(const _Float16* __restrict__ h2, const _Float16* __restrict__ sk,
                       const float* __restrict__ fin, _Float16* __restrict__ xc){
  size_t i = (size_t)blockIdx.x*256 + threadIdx.x;
  int c = (int)((i >> 9) & 63);
  float v = fin[c]*(float)h2[i] + fin[64 + c] + (float)sk[i];
  xc[i] = (_Float16)v;
}

// ------------------------------------------------------------------
// maxpool(3) + pred/att heads + softsign. One block per m.
__global__ __launch_bounds__(256) void kfinal(
    const _Float16* __restrict__ xc,
    const float* __restrict__ Wp, const float* __restrict__ bp,
    const float* __restrict__ Wat, const float* __restrict__ bat,
    const float* __restrict__ Wh, const float* __restrict__ bh,
    float* __restrict__ out)
{
  __shared__ float pooled[64*170];
  __shared__ float attw[170];
  __shared__ float wp[64];
  __shared__ float pa[48];
  __shared__ float scal[2];
  int tid = threadIdx.x, m = blockIdx.x;

  for (int i = tid; i < 64*170; i += 256){
    int c = i/170, tp = i - c*170;
    const _Float16* p = xc + ((size_t)m*64 + c)*512 + tp*3;
    pooled[i] = fmaxf((float)p[0], fmaxf((float)p[1], (float)p[2]));
  }
  __syncthreads();
  for (int tp = tid; tp < 170; tp += 256){
    float s = bat[0];
    for (int c = 0; c < 64; ++c) s += Wat[c]*pooled[c*170 + tp];
    attw[tp] = s;
  }
  __syncthreads();
  if (tid == 0){ float mx = -1e30f; for (int tp = 0; tp < 170; ++tp) mx = fmaxf(mx, attw[tp]); scal[0] = mx; }
  __syncthreads();
  for (int tp = tid; tp < 170; tp += 256) attw[tp] = __expf(attw[tp] - scal[0]);
  __syncthreads();
  if (tid == 0){ float z = 0.f; for (int tp = 0; tp < 170; ++tp) z += attw[tp]; scal[1] = 1.f/z; }
  __syncthreads();
  if (tid < 64){
    float s = 0.f;
    for (int tp = 0; tp < 170; ++tp) s += attw[tp]*pooled[tid*170 + tp];
    wp[tid] = s*scal[1];
  }
  __syncthreads();
  if (tid < 48){
    float s = bp[tid];
    for (int c = 0; c < 64; ++c) s += Wp[tid*64 + c]*wp[c];
    pa[tid] = s*(1.f/170.f);
  }
  __syncthreads();
  if (tid < 32){
    float s = bh[tid];
    for (int o = 0; o < 48; ++o) s += pa[o]*Wh[o*32 + tid];
    out[(size_t)m*32 + tid] = s/(1.f + fabsf(s));
  }
}

// ------------------------------------------------------------------
extern "C" void kernel_launch(void* const* d_in, const int* in_sizes, int n_in,
                              void* d_out, int out_size, void* d_ws, size_t ws_size,
                              hipStream_t stream) {
  (void)in_sizes; (void)n_in; (void)out_size; (void)ws_size;
  auto F = [&](int i){ return (const float*)d_in[i]; };
  char* ws = (char*)d_ws;

  // workspace layout (bytes)
  const size_t SZT = (size_t)1024*64*512*2;      // 64 MB per f16 tensor
  _Float16* xc  = (_Float16*)(ws + 0);
  _Float16* h1  = (_Float16*)(ws + SZT);
  _Float16* h2  = (_Float16*)(ws + 2*SZT);
  _Float16* skp = (_Float16*)(ws + 3*SZT);
  float* stbase = (float*)(ws + 4*SZT);          // 6 regions x 32768 floats
  float* fnbase = (float*)(ws + 4*SZT + 6*131072);
  _Float16* wz  = (_Float16*)(ws + 4*SZT + 6*131072 + 6*512);
  auto ST = [&](int i){ return stbase + (size_t)i*32768; };
  auto FN = [&](int i){ return fnbase + (size_t)i*128; };

  // swizzled weight slots (offsets in halfs)
  _Float16 *WG0a = wz +      0, *WS0a = wz +  12288, *WK0  = wz +  24576;
  _Float16 *WG0b = wz +  36864, *WS0b = wz +  57344;
  _Float16 *WG1a = wz +  77824, *WS1a = wz +  98304, *WK1  = wz + 118784;
  _Float16 *WG1b = wz + 139264, *WS1b = wz + 159744;
  _Float16 *WG2a = wz + 180224, *WS2a = wz + 200704, *WK2  = wz + 221184;
  _Float16 *WG2b = wz + 241664, *WS2b = wz + 262144;

  kzero<<<768, 256, 0, stream>>>(stbase, 6*32768);

  kprepw<36,false><<<48, 256, 0, stream>>>(F(15), WG0a);
  kprepw<36,false><<<48, 256, 0, stream>>>(F(17), WS0a);
  kprepw<36,true ><<<48, 256, 0, stream>>>(F(27), WK0);
  kprepw<64,false><<<80, 256, 0, stream>>>(F(19), WG0b);
  kprepw<64,false><<<80, 256, 0, stream>>>(F(21), WS0b);
  kprepw<64,false><<<80, 256, 0, stream>>>(F(29), WG1a);
  kprepw<64,false><<<80, 256, 0, stream>>>(F(31), WS1a);
  kprepw<64,true ><<<80, 256, 0, stream>>>(F(41), WK1);
  kprepw<64,false><<<80, 256, 0, stream>>>(F(33), WG1b);
  kprepw<64,false><<<80, 256, 0, stream>>>(F(35), WS1b);
  kprepw<64,false><<<80, 256, 0, stream>>>(F(43), WG2a);
  kprepw<64,false><<<80, 256, 0, stream>>>(F(45), WS2a);
  kprepw<64,true ><<<80, 256, 0, stream>>>(F(55), WK2);
  kprepw<64,false><<<80, 256, 0, stream>>>(F(47), WG2b);
  kprepw<64,false><<<80, 256, 0, stream>>>(F(49), WS2b);

  kgat<<<dim3(64,64), 256, 0, stream>>>(F(0),
      F(3),F(4),F(5),F(6),F(7),F(8),
      F(9),F(10),F(11),F(12),F(13),F(14), xc);

  dim3 cg(32, 1024);
  // ---- block 0 (cin=36, dil=1/2)
  kconv<36, 1, true, false><<<cg, 128, 0, stream>>>(xc, WG0a, WS0a, WK0, F(16), F(18), F(28), nullptr, h1, skp, ST(0));
  kbnfin<<<1, 64, 0, stream>>>(ST(0), F(23), F(24), FN(0));
  kconv<64, 2, false, true><<<cg, 128, 0, stream>>>(h1, WG0b, WS0b, nullptr, F(20), F(22), nullptr, FN(0), h2, nullptr, ST(1));
  kbnfin<<<1, 64, 0, stream>>>(ST(1), F(25), F(26), FN(1));
  kapply<<<131072, 256, 0, stream>>>(h2, skp, FN(1), xc);
  // ---- block 1 (dil=4/8)
  kconv<64, 4, true, false><<<cg, 128, 0, stream>>>(xc, WG1a, WS1a, WK1, F(30), F(32), F(42), nullptr, h1, skp, ST(2));
  kbnfin<<<1, 64, 0, stream>>>(ST(2), F(37), F(38), FN(2));
  kconv<64, 8, false, true><<<cg, 128, 0, stream>>>(h1, WG1b, WS1b, nullptr, F(34), F(36), nullptr, FN(2), h2, nullptr, ST(3));
  kbnfin<<<1, 64, 0, stream>>>(ST(3), F(39), F(40), FN(3));
  kapply<<<131072, 256, 0, stream>>>(h2, skp, FN(3), xc);
  // ---- block 2 (dil=16/32)
  kconv<64, 16, true, false><<<cg, 128, 0, stream>>>(xc, WG2a, WS2a, WK2, F(44), F(46), F(56), nullptr, h1, skp, ST(4));
  kbnfin<<<1, 64, 0, stream>>>(ST(4), F(51), F(52), FN(4));
  kconv<64, 32, false, true><<<cg, 128, 0, stream>>>(h1, WG2b, WS2b, nullptr, F(48), F(50), nullptr, FN(4), h2, nullptr, ST(5));
  kbnfin<<<1, 64, 0, stream>>>(ST(5), F(53), F(54), FN(5));
  kapply<<<131072, 256, 0, stream>>>(h2, skp, FN(5), xc);

  kfinal<<<1024, 256, 0, stream>>>(xc, F(57), F(58), F(59), F(60), F(61), F(62), (float*)d_out);
}
